// DVC_ChannelARpriors_34437047779595
// MI455X (gfx1250) — matrix operand order repaired and checked by reading, not compiled, not run on hardware
//
#include <hip/hip_runtime.h>
#include <hip/hip_bf16.h>
#include <stdint.h>

// ---------------------------------------------------------------------------
// MI455X (gfx1250) implementation of the DVC channel-AR compression forward.
// Core: implicit-GEMM conv/deconv on v_wmma_f32_16x16x32_f16 (wave32).
// Weights pre-converted to f16 once per layer; weight tiles staged into LDS
// by the Tensor Data Mover (tensor_load_to_lds, 2-byte elements). A and B
// WMMA operands are fetched as aligned ds_load_b128 vectors (no per-step
// conversions). LDS K-decode tables + hoisted pixel decode kill all per-step
// integer division. Bias + LeakyReLU fused into the GEMM epilogue.
// ---------------------------------------------------------------------------

typedef _Float16 v16h __attribute__((ext_vector_type(16)));
typedef _Float16 v8h  __attribute__((ext_vector_type(8)));
typedef float    v8f  __attribute__((ext_vector_type(8)));
typedef unsigned int       u32;
typedef unsigned long long u64;
typedef u32 u32x4 __attribute__((ext_vector_type(4)));
typedef int i32x4 __attribute__((ext_vector_type(4)));
typedef int i32x8 __attribute__((ext_vector_type(8)));

#define KMAX 9600  // max Cin*k*k in this network (384*25)

// =========================== conv implicit GEMM ============================

struct ConvArgs {
  const float*    x;    // input  (B,Cin,H,W), f32
  const _Float16* w;    // f16 weights: conv (Cout,Cin*k*k) row-major; deconv (Cin,Cout,k,k)
  const float*    bias; // (Cout)
  float*          y;    // output (B,Cout,Hout,Wout)
  int B, Cin, H, W, Cout, K, k, sshift, pad, Hout, Wout, NP;  // stride = 1<<sshift
  int fuse;        // 0 = bias only, 1 = bias + LeakyReLU(0.01)
  int transposed;  // ConvTranspose2d gather rule
  int use_tdm;     // stage weight tile via Tensor Data Mover
};

// Block tile: 64 (M=Cout) x 128 (N=pixels) x 32 (K). 8 waves, each 32x32.
__global__ __launch_bounds__(256)
void conv_igemm_wmma(ConvArgs a) {
  const int tid  = threadIdx.x;
  const int lane = tid & 31;
  const int wave = tid >> 5;
  const int mtile0 = blockIdx.y * 64;
  const int ntile0 = blockIdx.x * 128;
  const int wm = (wave >> 2) * 32;   // wave row offset inside block tile
  const int wn = (wave & 3) * 32;    // wave col offset inside block tile

  __shared__ __attribute__((aligned(64))) _Float16 shA[64 * 32];  // weight tile f16 (TDM dest, LDS base)
  __shared__ __attribute__((aligned(32))) _Float16 shB[8 * 512];  // B tile, fragment-order f16
  __shared__ short         sKci[KMAX];                            // k-index -> input channel
  __shared__ unsigned char sKd[KMAX];                             // k-index -> (kr<<3)|kc

  const int khw  = a.k * a.k;
  const int HoWo = a.Hout * a.Wout;
  const int HWin = a.H * a.W;

  // ---- one-time K decode table (replaces per-step div/mod storms) ----
  for (int t = tid; t < a.K; t += 256) {
    int ci = t / khw, rr = t - ci * khw;
    int kr = rr / a.k, kc = rr - kr * a.k;
    sKci[t] = (short)ci;
    sKd[t]  = (unsigned char)((kr << 3) | kc);
  }

  // ---- one-time pixel decode: this thread owns column nn for all K steps ----
  const int nn = tid & 127;          // output-pixel column within block tile
  const int kb = tid >> 7;           // K parity base: this thread covers kk = kb + 2r
  const int gn = ntile0 + nn;
  int pvalid = (gn < a.NP);
  int pb = 0, poh = 0, pow_ = 0;
  if (pvalid) {
    pb = gn / HoWo; int rem = gn - pb * HoWo;
    poh = rem / a.Wout; pow_ = rem - poh * a.Wout;
  }
  const u64 pixbase = (u64)pb * (u64)a.Cin * (u64)HWin;
  const int smask = (1 << a.sshift) - 1;

  v8f acc[2][2] = {};
  __syncthreads();  // K tables ready

  for (int k0 = 0; k0 < a.K; k0 += 32) {
    // ---------------- A tile: f16 weights [64 x 32] ----------------
    if (!a.transposed && a.use_tdm) {
      if (tid < 32) {  // wave 0 issues the TDM descriptor
        u64 gaddr = reinterpret_cast<u64>(a.w) + ((u64)mtile0 * (u64)a.K + (u64)k0) * 2ull;
        u32 td0 = (u32)(a.K - k0);          // remaining row length  -> OOB zero-fill
        u32 td1 = (u32)(a.Cout - mtile0);   // remaining rows        -> OOB zero-fill
        u64 st0 = (u64)a.K;                 // row stride in elements
        u32x4 g0;
        g0.x = 1u;                                    // count=1, no gather
        g0.y = 0u;                                    // lds_addr = base of shA
        g0.z = (u32)(gaddr & 0xFFFFFFFFull);
        g0.w = (u32)((gaddr >> 32) & 0x1FFFFFFull) | (2u << 30);  // type=2 (image)
        i32x8 g1;
        g1[0] = (int)(1u << 16);                      // data_size = 2B (f16)
        g1[1] = (int)((td0 & 0xFFFFu) << 16);         // tensor_dim0[15:0]
        g1[2] = (int)(((td0 >> 16) & 0xFFFFu) | ((td1 & 0xFFFFu) << 16));
        g1[3] = (int)(((td1 >> 16) & 0xFFFFu) | (32u << 16));     // tile_dim0 = 32
        g1[4] = (int)(64u);                           // tile_dim1 = 64, tile_dim2 = 0
        g1[5] = (int)(st0 & 0xFFFFFFFFull);           // tensor_dim0_stride lo
        g1[6] = (int)((st0 >> 32) & 0xFFFFull);       // tensor_dim0_stride hi
        g1[7] = 0;
        i32x4 g2 = {}; i32x4 g3 = {};
        i32x8 g4 = {};                                // extra group (6-arg toolchain form)
        __builtin_amdgcn_tensor_load_to_lds(g0, g1, g2, g3, g4, 0);
        __builtin_amdgcn_s_wait_tensorcnt(0);
      }
    } else {
      if (tid == 0)  // pull weight tile toward L2 (global_prefetch_b8)
        __builtin_prefetch(a.w + (u64)mtile0 * (u64)a.K + (u64)k0, 0, 1);
      #pragma unroll
      for (int r = 0; r < 8; ++r) {
        int idx = tid + r * 256;           // 0..2047
        int mm = idx >> 5, kk = idx & 31;
        int gm = mtile0 + mm, gk = k0 + kk;
        _Float16 val = (_Float16)0.f;
        if (gm < a.Cout && gk < a.K) {
          if (!a.transposed) {
            val = a.w[(u64)gm * (u64)a.K + gk];
          } else {
            int ci = sKci[gk], d = sKd[gk];
            val = a.w[(((u64)ci * a.Cout + gm) * a.k + (d >> 3)) * a.k + (d & 7)];
          }
        }
        shA[mm * 32 + kk] = val;
      }
    }

    // ------- B tile: im2col gather, stored in WMMA fragment order -------
    // B 32x16 layout (ISA B-matrix): lane = ((K>>3)&1)*16 + n, half index = (K&7) + (K>>4)*8.
    #pragma unroll
    for (int r = 0; r < 16; ++r) {
      int kk = kb + 2 * r;               // 0..31 across the two thread halves
      int gk = k0 + kk;
      float val = 0.f;
      if (pvalid && gk < a.K) {
        int ci = sKci[gk], d = sKd[gk];
        int kr = d >> 3, kc = d & 7;
        if (!a.transposed) {
          int ih = (poh << a.sshift) - a.pad + kr;
          int iw = (pow_ << a.sshift) - a.pad + kc;
          if ((unsigned)ih < (unsigned)a.H && (unsigned)iw < (unsigned)a.W)
            val = a.x[pixbase + (u64)ci * HWin + ih * a.W + iw];
        } else {  // ConvTranspose2d: oh = ih*s + kr - pad
          int ihs = poh + a.pad - kr, iws = pow_ + a.pad - kc;
          if (ihs >= 0 && iws >= 0 && ((ihs | iws) & smask) == 0) {
            int ih = ihs >> a.sshift, iw = iws >> a.sshift;
            if (ih < a.H && iw < a.W)
              val = a.x[pixbase + (u64)ci * HWin + ih * a.W + iw];
          }
        }
      }
      shB[(nn >> 4) * 512 + ((((kk >> 3) & 1) * 16 + (nn & 15)) << 4) + ((kk & 7) + ((kk >> 4) << 3))] = (_Float16)val;
    }
    __syncthreads();

    // ------------- fragments + WMMA (CDNA5 layouts, wave32) -------------
    // A 16x32 f16 per-lane operand = two contiguous 16B chunks of the f16 row:
    //   halves [kg*8, kg*8+8) -> VGPR0..3 and [16+kg*8, ...) -> VGPR4..7.
    const int fm = lane & 15, kg = lane >> 4;
    v16h afrag[2], bfrag[2];
    #pragma unroll
    for (int t = 0; t < 2; ++t) {
      const _Float16* Ar = &shA[(wm + t * 16 + fm) * 32 + (kg << 3)];
      v8h alo = *(const v8h*)(Ar);         // ds_load_b128
      v8h ahi = *(const v8h*)(Ar + 16);    // ds_load_b128
      afrag[t] = __builtin_shufflevector(alo, ahi, 0, 1, 2, 3, 4, 5, 6, 7,
                                                   8, 9, 10, 11, 12, 13, 14, 15);
      // B operand: one aligned 32-byte vector load per tile (2x ds_load_b128)
      bfrag[t] = *(const v16h*)&shB[((wn >> 4) + t) * 512 + lane * 16];
    }
    #pragma unroll
    for (int i = 0; i < 2; ++i)
      #pragma unroll
      for (int j = 0; j < 2; ++j)
        acc[i][j] = __builtin_amdgcn_wmma_f32_16x16x32_f16(
            false, afrag[i], false, bfrag[j], (short)0, acc[i][j], false, false);
    __syncthreads();
  }

  // ---------------- epilogue: bias (+leaky), NCHW scatter ----------------
  // C/D f32 16x16 layout: lane L -> col n=L&15, rows m = 8*(L>>4) + e
  const int cn = lane & 15, mrow0 = (lane >> 4) * 8;
  #pragma unroll
  for (int j = 0; j < 2; ++j) {
    int gcol = ntile0 + wn + j * 16 + cn;
    if (gcol >= a.NP) continue;
    int b = gcol / HoWo, rem = gcol - b * HoWo;
    #pragma unroll
    for (int i = 0; i < 2; ++i) {
      #pragma unroll
      for (int e = 0; e < 8; ++e) {
        int gm = mtile0 + wm + i * 16 + mrow0 + e;
        if (gm >= a.Cout) continue;
        float v = acc[i][j][e] + a.bias[gm];
        if (a.fuse == 1) v = (v >= 0.f) ? v : 0.01f * v;
        a.y[((u64)b * a.Cout + gm) * HoWo + rem] = v;
      }
    }
  }
}

// =================== weight f32 -> f16 conversion pass =====================

__global__ __launch_bounds__(256)
void cvt_f16_kernel(const float* src, _Float16* dst, long long n) {
  long long i = (long long)blockIdx.x * 256 + threadIdx.x;
  if (i < n) dst[i] = (_Float16)src[i];
}

// =========================== elementwise kernel ============================

struct EwArgs {
  float* out; const float* a; const float* b; const float* c; const float* chan;
  long long n; int C; int HW; int mode;
};

__device__ __forceinline__ float sigmoidf_(float x) { return 1.f / (1.f + expf(-x)); }

__global__ __launch_bounds__(256) void ew_kernel(EwArgs e) {
  long long i = (long long)blockIdx.x * 256 + threadIdx.x;
  if (i >= e.n) return;
  float r = 0.f;
  switch (e.mode) {
    case 0: { float v = e.a[i]; r = v * v; } break;                        // square
    case 1: r = e.a[i] * rsqrtf(fmaxf(e.b[i], 1e-9f)); break;              // GDN
    case 2: r = e.a[i] * sqrtf(fmaxf(e.b[i], 0.f)); break;                 // IGDN
    case 3: { float v = e.a[i]; r = v >= 0.f ? v : 0.01f * v; } break;     // leaky
    case 4: { float m = e.b[i]; r = roundf(e.a[i] - m) + m; } break;       // STE round vs mean
    case 5: { int ch = (int)((i / e.HW) % e.C); float m = e.chan[ch];
              r = roundf(e.a[i] - m) + m; } break;                         // round vs median
    case 6: { float s = fmaxf(e.b[i], 0.11f);                              // gaussian likelihood
              float v = fabsf(e.a[i] - e.c[i]);
              const float is2 = 0.70710678118654752440f;
              float up = 0.5f * erfcf((v - 0.5f) * is2 / s);
              float lo = 0.5f * erfcf((v + 0.5f) * is2 / s);
              r = fmaxf(up - lo, 1e-9f); } break;
    case 7: r = e.a[i] + 0.5f * tanhf(e.b[i]); break;                      // LRP add
    default: r = e.a[i]; break;
  }
  e.out[i] = r;
}

// channel-slab copy: src(B,Csrc,HW)[soff..soff+C) -> dst(B,Cdst,HW)[doff..)
__global__ __launch_bounds__(256)
void copy_ch_kernel(const float* src, int Csrc, int soff, float* dst, int Cdst,
                    int doff, int B, int C, int HW) {
  long long i = (long long)blockIdx.x * 256 + threadIdx.x;
  long long tot = (long long)B * C * HW;
  if (i >= tot) return;
  int hw = (int)(i % HW); long long t = i / HW;
  int c = (int)(t % C);   int b = (int)(t / C);
  dst[((u64)b * Cdst + (doff + c)) * HW + hw] =
      src[((u64)b * Csrc + (soff + c)) * HW + hw];
}

// =========================== entropy bottleneck ============================

struct EbArgs {
  const float* z; float* lik; const float* med;
  const float* m0; const float* m1; const float* m2; const float* m3; const float* m4;
  const float* b0; const float* b1; const float* b2; const float* b3; const float* b4;
  const float* f0; const float* f1; const float* f2; const float* f3;
  long long n; int C; int HW;
};

__device__ __forceinline__ float softplusf_(float x) {
  return x > 20.f ? x : log1pf(expf(x));
}

__device__ float eb_logits(const EbArgs& e, int c, float v) {
  float h[3], h2[3];
  #pragma unroll
  for (int o = 0; o < 3; ++o) {
    h[o] = softplusf_(e.m0[c * 3 + o]) * v + e.b0[c * 3 + o];
    h[o] += tanhf(e.f0[c * 3 + o]) * tanhf(h[o]);
  }
  const float* Ms[3] = {e.m1, e.m2, e.m3};
  const float* Bs[3] = {e.b1, e.b2, e.b3};
  const float* Fs[3] = {e.f1, e.f2, e.f3};
  #pragma unroll
  for (int L = 0; L < 3; ++L) {
    #pragma unroll
    for (int o = 0; o < 3; ++o) {
      float s = Bs[L][c * 3 + o];
      #pragma unroll
      for (int i = 0; i < 3; ++i) s += softplusf_(Ms[L][(c * 3 + o) * 3 + i]) * h[i];
      h2[o] = s;
    }
    #pragma unroll
    for (int o = 0; o < 3; ++o)
      h[o] = h2[o] + tanhf(Fs[L][c * 3 + o]) * tanhf(h2[o]);
  }
  float s = e.b4[c];
  #pragma unroll
  for (int i = 0; i < 3; ++i) s += softplusf_(e.m4[c * 3 + i]) * h[i];
  return s;
}

__global__ __launch_bounds__(256) void eb_kernel(EbArgs e) {
  long long i = (long long)blockIdx.x * 256 + threadIdx.x;
  if (i >= e.n) return;
  int c = (int)((i / e.HW) % e.C);
  float v = e.z[i];  // already round(z - med) + med
  float lo = eb_logits(e, c, v - 0.5f);
  float up = eb_logits(e, c, v + 0.5f);
  float sg = (lo + up) > 0.f ? -1.f : 1.f;
  float lik = fabsf(sigmoidf_(sg * up) - sigmoidf_(sg * lo));
  e.lik[i] = fmaxf(lik, 1e-9f);
}

// ================================ host =====================================

struct CP { const float* b; const float* w; };
struct GP { const float* beta; const float* gamma; };

extern "C" void kernel_launch(void* const* d_in, const int* in_sizes, int n_in,
                              void* d_out, int out_size, void* d_ws, size_t ws_size,
                              hipStream_t stream) {
  (void)in_sizes; (void)out_size; (void)ws_size;
  const float* x    = (const float*)d_in[0];
  const float* side = (const float*)d_in[1];

  // -------- walk params pytree (sorted dict keys, {b,w} / {beta,gamma}) ----
  int pi = 2;
  auto nf = [&]() -> const float* {
    return (pi < n_in) ? (const float*)d_in[pi++] : nullptr;
  };
  CP ccm[8][3], ccs[8][3], lr[8][3];
  for (int i = 0; i < 8; ++i) for (int j = 0; j < 3; ++j) { ccm[i][j].b = nf(); ccm[i][j].w = nf(); }
  for (int i = 0; i < 8; ++i) for (int j = 0; j < 3; ++j) { ccs[i][j].b = nf(); ccs[i][j].w = nf(); }
  CP dec[4]; GP decg[3];
  for (int l = 0; l < 4; ++l) {
    dec[l].b = nf(); dec[l].w = nf();
    if (l < 3) { decg[l].beta = nf(); decg[l].gamma = nf(); }
  }
  const float *eb_b[5], *eb_f[4], *eb_m[5], *eb_med;
  for (int i = 0; i < 5; ++i) eb_b[i] = nf();
  for (int i = 0; i < 4; ++i) eb_f[i] = nf();
  for (int i = 0; i < 5; ++i) eb_m[i] = nf();
  eb_med = nf();
  CP encs[4]; GP encsg[3];
  for (int l = 0; l < 4; ++l) {
    encs[l].b = nf(); encs[l].w = nf();
    if (l < 3) { encsg[l].beta = nf(); encsg[l].gamma = nf(); }
  }
  CP encx[4]; GP encxg[3];
  for (int l = 0; l < 4; ++l) {
    encx[l].b = nf(); encx[l].w = nf();
    if (l < 3) { encxg[l].beta = nf(); encxg[l].gamma = nf(); }
  }
  CP ha[3]; for (int l = 0; l < 3; ++l) { ha[l].b = nf(); ha[l].w = nf(); }
  CP hs[3]; for (int l = 0; l < 3; ++l) { hs[l].b = nf(); hs[l].w = nf(); }
  for (int i = 0; i < 8; ++i) for (int j = 0; j < 3; ++j) { lr[i][j].b = nf(); lr[i][j].w = nf(); }

  // --------------------------- workspace carve -----------------------------
  float* ws = (float*)d_ws;
  u64 off = 0;
  auto alloc = [&](u64 nelem) { float* p = ws + off; off += (nelem + 255ull) & ~255ull; return p; };
  const u64 BIG = 4ull * 192 * 128 * 128;           // 12.58M floats
  float* t0 = alloc(BIG); float* t1 = alloc(BIG); float* t2 = alloc(BIG);
  float* y   = alloc(196608); float* yh   = alloc(196608);
  float* lm  = alloc(196608); float* ls   = alloc(196608);
  float* fs  = alloc(196608); float* yhS  = alloc(196608);
  float* lms = alloc(196608); float* lss  = alloc(196608);
  float* z   = alloc(12288);  float* zh   = alloc(12288);
  float* zs  = alloc(12288);  float* zsh  = alloc(12288);
  float* hbA = alloc(49152);  float* hbB  = alloc(294912); float* hbC = alloc(393216);
  float* sup = alloc(368640); float* sup3 = alloc(393216);
  float* st1 = alloc(131072); float* st2  = alloc(65536);
  float* mu  = alloc(24576);  float* sc   = alloc(24576);
  float* ysb = alloc(24576);  float* yqb  = alloc(24576);  float* lk24 = alloc(24576);
  float* cat = alloc(393216);
  _Float16* whbuf = (_Float16*)alloc(1048576);      // 2M halves: largest weight = 1.84M

  float* out_xhat = (float*)d_out;
  float* out_ylik = out_xhat + 786432;
  float* out_zlik = out_ylik + 196608;
  float* out_slik = out_zlik + 12288;

  // ------------------------------ launchers --------------------------------
  auto conv = [&](const float* in, CP p, float* outp, int B, int Cin, int H, int W,
                  int Cout, int k, int s, int fuse, int transposed) {
    long long wcnt = (long long)Cout * Cin * k * k;
    cvt_f16_kernel<<<dim3((unsigned)((wcnt + 255) / 256)), 256, 0, stream>>>(p.w, whbuf, wcnt);
    ConvArgs a;
    a.x = in; a.w = whbuf; a.bias = p.b; a.y = outp;
    a.B = B; a.Cin = Cin; a.H = H; a.W = W; a.Cout = Cout;
    a.k = k; a.sshift = (s == 2) ? 1 : 0; a.pad = k / 2;
    a.Hout = transposed ? H * s : (H + 2 * a.pad - k) / s + 1;
    a.Wout = transposed ? W * s : (W + 2 * a.pad - k) / s + 1;
    a.K = Cin * k * k; a.NP = B * a.Hout * a.Wout;
    a.fuse = fuse; a.transposed = transposed;
    a.use_tdm = transposed ? 0 : 1;
    dim3 grid((a.NP + 127) / 128, (a.Cout + 63) / 64);
    conv_igemm_wmma<<<grid, 256, 0, stream>>>(a);
  };
  auto ew = [&](int mode, float* outp, const float* A, const float* Bp,
                const float* Cc, const float* chan, long long n, int C, int HW) {
    EwArgs e{outp, A, Bp, Cc, chan, n, C, HW, mode};
    ew_kernel<<<dim3((unsigned)((n + 255) / 256)), 256, 0, stream>>>(e);
  };
  auto cpch = [&](const float* src, int Csrc, int soff, float* dst, int Cdst,
                  int doff, int B, int C, int HW) {
    long long n = (long long)B * C * HW;
    copy_ch_kernel<<<dim3((unsigned)((n + 255) / 256)), 256, 0, stream>>>(
        src, Csrc, soff, dst, Cdst, doff, B, C, HW);
  };
  auto gdn = [&](float* xa, GP g, float* tmp, float* norm, int B, int C, int H, int W, int inv) {
    long long n = (long long)B * C * H * W;
    ew(0, tmp, xa, nullptr, nullptr, nullptr, n, C, H * W);      // x^2
    CP gp{g.beta, g.gamma};                                      // 1x1 conv, bias=beta
    conv(tmp, gp, norm, B, C, H, W, C, 1, 1, 0, 0);
    ew(inv ? 2 : 1, xa, xa, norm, nullptr, nullptr, n, C, H * W);
  };
  auto analysisF = [&](const float* in, CP* cp, GP* gp, float* outY) {
    conv(in, cp[0], t0, 4, 3, 256, 256, 192, 5, 2, 0, 0);        // ->128x128
    gdn(t0, gp[0], t1, t2, 4, 192, 128, 128, 0);
    conv(t0, cp[1], t1, 4, 192, 128, 128, 192, 5, 2, 0, 0);      // ->64x64
    gdn(t1, gp[1], t0, t2, 4, 192, 64, 64, 0);
    conv(t1, cp[2], t0, 4, 192, 64, 64, 192, 5, 2, 0, 0);        // ->32x32
    gdn(t0, gp[2], t1, t2, 4, 192, 32, 32, 0);
    conv(t0, cp[3], outY, 4, 192, 32, 32, 192, 5, 2, 0, 0);      // ->16x16 (M=192)
  };
  auto hyperA = [&](const float* in, float* outZ) {
    conv(in, ha[0], hbB, 4, 192, 16, 16, 192, 3, 1, 1, 0);
    conv(hbB, ha[1], hbA, 4, 192, 16, 16, 192, 5, 2, 1, 0);      // ->8x8
    conv(hbA, ha[2], outZ, 4, 192, 8, 8, 192, 5, 2, 0, 0);       // ->4x4
  };
  auto hyperS = [&](const float* zhat, float* olm, float* ols) {
    conv(zhat, hs[0], hbA, 4, 192, 4, 4, 192, 5, 2, 1, 1);       // deconv ->8x8
    conv(hbA, hs[1], hbB, 4, 192, 8, 8, 288, 5, 2, 1, 1);        // deconv ->16x16
    conv(hbB, hs[2], hbC, 4, 288, 16, 16, 384, 3, 1, 0, 0);      // (4,384,16,16)
    cpch(hbC, 384, 0,   olm, 192, 0, 4, 192, 256);
    cpch(hbC, 384, 192, ols, 192, 0, 4, 192, 256);
  };
  auto seq3 = [&](const float* in, int Cin, CP* p, int c1, int c2, float* outp) {
    conv(in, p[0], st1, 4, Cin, 16, 16, c1, 3, 1, 1, 0);
    conv(st1, p[1], st2, 4, c1, 16, 16, c2, 3, 1, 1, 0);
    conv(st2, p[2], outp, 4, c2, 16, 16, 24, 3, 1, 0, 0);
  };
  auto channelAR = [&](const float* yin, const float* mlm, const float* mls,
                       float* yhat, float* likOut, CP (*pm)[3], CP (*ps)[3], CP (*pl)[3]) {
    const int HW = 256;
    const long long n24 = 4ll * 24 * HW;
    for (int i = 0; i < 8; ++i) {
      int Cs = 192 + 24 * i;
      cpch(mlm, 192, 0, sup, Cs, 0, 4, 192, HW);                 // mean support
      if (i) cpch(yhat, 192, 0, sup, Cs, 192, 4, 24 * i, HW);
      seq3(sup, Cs, pm[i], 64 + 8 * i, 32 + 4 * i, mu);
      cpch(mls, 192, 0, sup, Cs, 0, 4, 192, HW);                 // scale support
      if (i) cpch(yhat, 192, 0, sup, Cs, 192, 4, 24 * i, HW);
      seq3(sup, Cs, ps[i], 64 + 8 * i, 32 + 4 * i, sc);
      cpch(yin, 192, 24 * i, ysb, 24, 0, 4, 24, HW);             // y slice
      ew(4, yqb, ysb, mu, nullptr, nullptr, n24, 24, HW);        // round(ys-mu)+mu
      ew(6, lk24, yqb, sc, mu, nullptr, n24, 24, HW);            // gaussian likelihood
      cpch(lk24, 24, 0, likOut, 192, 24 * i, 4, 24, HW);
      int Cs3 = Cs + 24;                                         // lrp support
      cpch(mlm, 192, 0, sup3, Cs3, 0, 4, 192, HW);
      if (i) cpch(yhat, 192, 0, sup3, Cs3, 192, 4, 24 * i, HW);
      cpch(yqb, 24, 0, sup3, Cs3, Cs, 4, 24, HW);
      seq3(sup3, Cs3, pl[i], 64 + 8 * (i + 1), 32 + 4 * (i + 1), mu);
      ew(7, yqb, yqb, mu, nullptr, nullptr, n24, 24, HW);        // yh += 0.5*tanh(lrp)
      cpch(yqb, 24, 0, yhat, 192, 24 * i, 4, 24, HW);
    }
  };

  // ------------------------------- pipeline --------------------------------
  analysisF(x, encx, encxg, y);                                  // y = g_a(x)
  hyperA(y, z);                                                  // z = h_a(y)
  ew(5, zh, z, nullptr, nullptr, eb_med, 12288, 192, 16);        // z_hat
  {
    EbArgs e{zh, out_zlik, eb_med,
             eb_m[0], eb_m[1], eb_m[2], eb_m[3], eb_m[4],
             eb_b[0], eb_b[1], eb_b[2], eb_b[3], eb_b[4],
             eb_f[0], eb_f[1], eb_f[2], eb_f[3], 12288ll, 192, 16};
    eb_kernel<<<dim3((12288 + 255) / 256), 256, 0, stream>>>(e);
  }
  hyperS(zh, lm, ls);
  channelAR(y, lm, ls, yh, out_ylik, ccm, ccs, lr);

  analysisF(side, encs, encsg, fs);                              // side branch
  hyperA(fs, zs);
  ew(5, zsh, zs, nullptr, nullptr, eb_med, 12288, 192, 16);
  hyperS(zsh, lms, lss);
  channelAR(fs, lms, lss, yhS, out_slik, ccm, ccs, lr);

  cpch(yh, 192, 0, cat, 384, 0,   4, 192, 256);                  // concat [y_hat, fea_side]
  cpch(fs, 192, 0, cat, 384, 192, 4, 192, 256);
  conv(cat, dec[0], t0, 4, 384, 16, 16, 192, 5, 2, 0, 1);        // synthesis
  gdn(t0, decg[0], t1, t2, 4, 192, 32, 32, 1);
  conv(t0, dec[1], t1, 4, 192, 32, 32, 192, 5, 2, 0, 1);
  gdn(t1, decg[1], t0, t2, 4, 192, 64, 64, 1);
  conv(t1, dec[2], t0, 4, 192, 64, 64, 192, 5, 2, 0, 1);
  gdn(t0, decg[2], t1, t2, 4, 192, 128, 128, 1);
  conv(t0, dec[3], out_xhat, 4, 192, 128, 128, 3, 5, 2, 0, 1);   // x_hat (4,3,256,256)
}
